// MHA_47201690583607
// MI455X (gfx1250) — matrix operand order repaired and checked by reading, not compiled, not run on hardware
//
#include <hip/hip_runtime.h>
#include <hip/hip_bf16.h>

typedef __attribute__((ext_vector_type(16))) __bf16 v16bf;
typedef __attribute__((ext_vector_type(8)))  __bf16 v8bf;
typedef __attribute__((ext_vector_type(8)))  float  v8f;

#define EMBED 1024
#define HEADS 16
#define HDIM  64
#define SEQ   2048
#define BATCH 4
#define MROWS (BATCH * SEQ)   // 8192

#if __has_builtin(__builtin_amdgcn_global_load_async_to_lds_b128) && \
    __has_builtin(__builtin_amdgcn_s_wait_asynccnt)
#define USE_ASYNC 1
#define ASYNC_WAIT(n) __builtin_amdgcn_s_wait_asynccnt(n)
#else
#define USE_ASYNC 0
#define ASYNC_WAIT(n) do { } while (0)
#endif

// types matching the async-copy builtin signature:
// (v4i addrspace(1)* src, v4i addrspace(3)* dst, imm offset, imm cpol)
typedef __attribute__((__vector_size__(16))) int gv4i;
typedef __attribute__((address_space(1))) gv4i* gv4i_glb;
typedef __attribute__((address_space(3))) gv4i* gv4i_lds;

// fp32 -> bf16 round-to-nearest-even
__device__ __forceinline__ unsigned short f2bf(float f) {
    union { float f; unsigned u; } x; x.f = f;
    unsigned u = x.u;
    unsigned r = (u + 0x7FFFu + ((u >> 16) & 1u)) >> 16;
    return (unsigned short)r;
}

// async (or sync fallback) 16-byte global -> LDS copy
__device__ __forceinline__ void g2l16(const unsigned short* g, unsigned short* l) {
#if USE_ASYNC
    __builtin_amdgcn_global_load_async_to_lds_b128((gv4i_glb)g, (gv4i_lds)l, 0, 0);
#else
    *(uint4*)l = *(const uint4*)g;
#endif
}

// Build a 16-element bf16 fragment from two 16-byte chunks.
__device__ __forceinline__ v16bf ldfrag(const unsigned short* p0, const unsigned short* p1) {
    v8bf lo = *(const v8bf*)p0;
    v8bf hi = *(const v8bf*)p1;
    return __builtin_shufflevector(lo, hi, 0,1,2,3,4,5,6,7,8,9,10,11,12,13,14,15);
}

// ---------------- conversion kernels ----------------
__global__ __launch_bounds__(256) void cvt_f32_bf16(const float* __restrict__ in,
                                                    unsigned short* __restrict__ out, int n) {
    int i = blockIdx.x * 256 + threadIdx.x;
    if (i < n) out[i] = f2bf(in[i]);
}

// W [Kdim x Ndim] fp32 -> Wt [Ndim x Kdim] bf16
__global__ __launch_bounds__(256) void cvt_transpose_bf16(const float* __restrict__ W,
                                                          unsigned short* __restrict__ Wt,
                                                          int Kdim, int Ndim) {
    int i = blockIdx.x * 256 + threadIdx.x;
    if (i < Kdim * Ndim) {
        int k = i / Ndim, n = i - k * Ndim;
        Wt[(size_t)n * Kdim + k] = f2bf(W[i]);
    }
}

// load one 128x32 A tile slice + 128x32 B tile slice into a pipeline stage
__device__ __forceinline__ void stage_load_gemm(unsigned short (*sA_)[40],
                                                unsigned short (*sB_)[40],
                                                const unsigned short* __restrict__ A,
                                                const unsigned short* __restrict__ Bt,
                                                int rowA0, int rowB0, int K, int kt, int tid) {
#pragma unroll
    for (int i = 0; i < 2; i++) {
        int c = tid + i * 256;
        int r = c >> 2, kc = (c & 3) * 8;
        g2l16(A  + (size_t)(rowA0 + r) * K + kt + kc, &sA_[r][kc]);
        g2l16(Bt + (size_t)(rowB0 + r) * K + kt + kc, &sB_[r][kc]);
    }
}

// ---------------- bf16 WMMA GEMM (3-stage async LDS pipeline) ----------------
// C[M,N] = A[M,K] * Bt[N,K]^T + bias
// EPI==0: scatter QKV epilogue (scale folded into Q, V stored transposed)
// EPI==1: fp32 output to Cout
template <int EPI>
__global__ __launch_bounds__(256)
void gemm_bf16(const unsigned short* __restrict__ A,
               const unsigned short* __restrict__ Bt,
               const float* __restrict__ bias,
               float* __restrict__ Cout,
               unsigned short* __restrict__ Qb,
               unsigned short* __restrict__ Kb,
               unsigned short* __restrict__ Vtb,
               int M, int N, int K) {
    __shared__ unsigned short sA[3][128][40];   // 3-stage, 128 rows x 32 k (pad 8)
    __shared__ unsigned short sB[3][128][40];

    const int bn = blockIdx.x, bm = blockIdx.y;
    const int tid = threadIdx.x;
    const int lane = tid & 31, wid = tid >> 5;
    const int wm = wid & 3, wn = wid >> 2;        // 4x2 wave grid -> 32x64 per wave
    const int l16 = lane & 15, half = lane >> 4;

    v8f acc[2][4];
#pragma unroll
    for (int mi = 0; mi < 2; mi++)
#pragma unroll
        for (int ni = 0; ni < 4; ni++)
#pragma unroll
            for (int v = 0; v < 8; v++) acc[mi][ni][v] = 0.f;

    const int rowA0 = bm * 128;
    const int rowB0 = bn * 128;
    const int nk = K >> 5;                        // 32-wide k tiles

    // prologue: stages 0 and 1 in flight (per wave: 4 async instr per stage)
    stage_load_gemm(sA[0], sB[0], A, Bt, rowA0, rowB0, K, 0, tid);
    stage_load_gemm(sA[1], sB[1], A, Bt, rowA0, rowB0, K, 32, tid);

    for (int i = 0; i < nk; i++) {
        const int st = i % 3;
        __syncthreads();            // everyone done reading stage (i-1) buffers
        if (i + 2 < nk) {
            stage_load_gemm(sA[(i + 2) % 3], sB[(i + 2) % 3], A, Bt,
                            rowA0, rowB0, K, (i + 2) * 32, tid);
            ASYNC_WAIT(8);          // stages i+1, i+2 still in flight; stage i landed
        } else if (i + 1 < nk) {
            ASYNC_WAIT(4);
        } else {
            ASYNC_WAIT(0);
        }
        __syncthreads();            // all waves' stage-i data visible in LDS

        v16bf af[2], bfr[4];
        const int ka = half * 8;    // A-frag k chunks: ka, ka+16
        const int kb = half * 8;    // B-frag k chunks: kb, kb+16
#pragma unroll
        for (int mi = 0; mi < 2; mi++) {
            int r = wm * 32 + mi * 16 + l16;
            af[mi] = ldfrag(&sA[st][r][ka], &sA[st][r][ka + 16]);
        }
#pragma unroll
        for (int ni = 0; ni < 4; ni++) {
            int r = wn * 64 + ni * 16 + l16;
            bfr[ni] = ldfrag(&sB[st][r][kb], &sB[st][r][kb + 16]);
        }
#pragma unroll
        for (int mi = 0; mi < 2; mi++)
#pragma unroll
            for (int ni = 0; ni < 4; ni++)
                acc[mi][ni] = __builtin_amdgcn_wmma_f32_16x16x32_bf16(
                    false, af[mi], false, bfr[ni], (short)0, acc[mi][ni], false, false);
    }

    // epilogue
#pragma unroll
    for (int mi = 0; mi < 2; mi++)
#pragma unroll
        for (int ni = 0; ni < 4; ni++)
#pragma unroll
            for (int v = 0; v < 8; v++) {
                int rl = wm * 32 + mi * 16 + v + 8 * half;
                int cl = wn * 64 + ni * 16 + l16;
                int gr = bm * 128 + rl;
                int gc = bn * 128 + cl;
                float val = acc[mi][ni][v] + bias[gc];
                if (EPI == 0) {
                    int sec = gc >> 10, cc = gc & 1023;
                    int h = cc >> 6, d = cc & 63;
                    int b = gr >> 11, s = gr & 2047;
                    size_t bh = (size_t)(b * HEADS + h);
                    if (sec == 0)
                        Qb[(bh * SEQ + s) * HDIM + d] = f2bf(val * 0.125f);  // 1/sqrt(64)
                    else if (sec == 1)
                        Kb[(bh * SEQ + s) * HDIM + d] = f2bf(val);
                    else
                        Vtb[(bh * HDIM + d) * SEQ + s] = f2bf(val);          // V transposed
                } else {
                    Cout[(size_t)gr * N + gc] = val;
                }
            }
}

// ---------------- flash attention ----------------
// Q,K: [B*H, S, D] bf16 (Q pre-scaled); Vt: [B*H, D, S] bf16; O: [B, S, E] bf16
__global__ __launch_bounds__(128)
void attn_kernel(const unsigned short* __restrict__ Q,
                 const unsigned short* __restrict__ Kg,
                 const unsigned short* __restrict__ Vt,
                 unsigned short* __restrict__ O) {
    __shared__ unsigned short sK[128][72];       // 128 keys x 64 d (pad 8)
    __shared__ unsigned short sV[64][136];       // 64 d x 128 t (pad 8)
    __shared__ unsigned short sP[4][16][136];    // per-wave P tile 16 x 128

    const int tid = threadIdx.x;
    const int lane = tid & 31, wid = tid >> 5;
    const int l16 = lane & 15, half = lane >> 4;
    const int bh = blockIdx.y;
    const int b = bh >> 4, h = bh & 15;
    const int s0 = blockIdx.x * 64;              // 4 waves x 16 rows

    // Q fragments in registers: rows s0+wid*16 .. +15, D=64 -> 2 k-steps
    const unsigned short* qrow = Q + ((size_t)bh * SEQ + s0 + wid * 16 + l16) * HDIM;
    v16bf aq[2];
#pragma unroll
    for (int j = 0; j < 2; j++)
        aq[j] = ldfrag(qrow + j * 32 + half * 8, qrow + j * 32 + half * 8 + 16);

    v8f oacc[4];
    float mrow[8], lrow[8];
#pragma unroll
    for (int nt = 0; nt < 4; nt++)
#pragma unroll
        for (int v = 0; v < 8; v++) oacc[nt][v] = 0.f;
#pragma unroll
    for (int v = 0; v < 8; v++) { mrow[v] = -3.0e38f; lrow[v] = 0.f; }

    for (int t0 = 0; t0 < SEQ; t0 += 128) {
        __syncthreads();
        // K tile: 128 x 64 bf16 = 1024 x 16B chunks, 8 per thread
#pragma unroll
        for (int i = 0; i < 8; i++) {
            int c = tid + i * 128;
            int r = c >> 3, kc = (c & 7) * 8;
            g2l16(Kg + ((size_t)bh * SEQ + t0 + r) * HDIM + kc, &sK[r][kc]);
        }
        // V^T tile: 64 x 128 bf16 = 1024 x 16B chunks
#pragma unroll
        for (int i = 0; i < 8; i++) {
            int c = tid + i * 128;
            int r = c >> 4, tc = (c & 15) * 8;
            g2l16(Vt + ((size_t)bh * HDIM + r) * SEQ + t0 + tc, &sV[r][tc]);
        }
        ASYNC_WAIT(0);
        __syncthreads();

        // S = Q @ K^T : 8 tiles along t
        v8f sc[8];
#pragma unroll
        for (int tt = 0; tt < 8; tt++) {
            v8f c;
#pragma unroll
            for (int v = 0; v < 8; v++) c[v] = 0.f;
#pragma unroll
            for (int ks = 0; ks < 2; ks++) {
                int rt = tt * 16 + l16;
                int dofs = half * 8 + ks * 32;
                v16bf bk = ldfrag(&sK[rt][dofs], &sK[rt][dofs + 16]);
                c = __builtin_amdgcn_wmma_f32_16x16x32_bf16(
                    false, aq[ks], false, bk, (short)0, c, false, false);
            }
            sc[tt] = c;
        }

        // online softmax (row = v + 8*half, replicated across 16 lanes)
        float tmax[8];
#pragma unroll
        for (int v = 0; v < 8; v++) {
            float m = sc[0][v];
#pragma unroll
            for (int tt = 1; tt < 8; tt++) m = fmaxf(m, sc[tt][v]);
            tmax[v] = m;
        }
#pragma unroll
        for (int off = 1; off < 16; off <<= 1)
#pragma unroll
            for (int v = 0; v < 8; v++)
                tmax[v] = fmaxf(tmax[v], __shfl_xor(tmax[v], off, 32));

        float alpha[8];
#pragma unroll
        for (int v = 0; v < 8; v++) {
            float mn = fmaxf(mrow[v], tmax[v]);
            alpha[v] = __expf(mrow[v] - mn);
            mrow[v] = mn;
        }
        float rsum[8];
#pragma unroll
        for (int v = 0; v < 8; v++) rsum[v] = 0.f;
#pragma unroll
        for (int tt = 0; tt < 8; tt++)
#pragma unroll
            for (int v = 0; v < 8; v++) {
                float p = __expf(sc[tt][v] - mrow[v]);
                sc[tt][v] = p;
                rsum[v] += p;
            }
#pragma unroll
        for (int off = 1; off < 16; off <<= 1)
#pragma unroll
            for (int v = 0; v < 8; v++)
                rsum[v] += __shfl_xor(rsum[v], off, 32);
#pragma unroll
        for (int v = 0; v < 8; v++) lrow[v] = lrow[v] * alpha[v] + rsum[v];
#pragma unroll
        for (int nt = 0; nt < 4; nt++)
#pragma unroll
            for (int v = 0; v < 8; v++) oacc[nt][v] *= alpha[v];

        // P (C-layout fp32) -> per-wave LDS bf16 tile (A layout source)
#pragma unroll
        for (int tt = 0; tt < 8; tt++)
#pragma unroll
            for (int v = 0; v < 8; v++)
                sP[wid][v + 8 * half][tt * 16 + l16] = f2bf(sc[tt][v]);
        __builtin_amdgcn_wave_barrier();   // per-wave tile; DS ops in-order within wave

        // O += P @ V : 4 k-steps (t) x 4 n-tiles (d)
#pragma unroll
        for (int ks = 0; ks < 4; ks++) {
            int kb2 = ks * 32 + half * 8;
            v16bf ap = ldfrag(&sP[wid][l16][kb2], &sP[wid][l16][kb2 + 16]);
#pragma unroll
            for (int nt = 0; nt < 4; nt++) {
                int rd = nt * 16 + l16;
                int tofs = half * 8 + ks * 32;
                v16bf bv = ldfrag(&sV[rd][tofs], &sV[rd][tofs + 16]);
                oacc[nt] = __builtin_amdgcn_wmma_f32_16x16x32_bf16(
                    false, ap, false, bv, (short)0, oacc[nt], false, false);
            }
        }
    }

    // finalize: O / l, store bf16 in [B,S,E] layout for the out-proj GEMM
#pragma unroll
    for (int v = 0; v < 8; v++) {
        float inv = 1.f / lrow[v];
        int gs = s0 + wid * 16 + v + 8 * half;
#pragma unroll
        for (int nt = 0; nt < 4; nt++) {
            int d = nt * 16 + l16;
            O[((size_t)b * SEQ + gs) * EMBED + h * HDIM + d] = f2bf(oacc[nt][v] * inv);
        }
    }
}

// ---------------- launch ----------------
extern "C" void kernel_launch(void* const* d_in, const int* in_sizes, int n_in,
                              void* d_out, int out_size, void* d_ws, size_t ws_size,
                              hipStream_t stream) {
    const float* x    = (const float*)d_in[0];   // [4,2048,1024]
    const float* Wqkv = (const float*)d_in[1];   // [1024,3072]
    const float* bqkv = (const float*)d_in[2];   // [3072]
    const float* Wout = (const float*)d_in[3];   // [1024,1024]
    const float* bout = (const float*)d_in[4];   // [1024]
    float* out = (float*)d_out;                  // [4,2048,1024] fp32

    char* ws = (char*)d_ws;
    size_t off = 0;
    auto carve = [&](size_t bytes) {
        void* p = ws + off;
        off = (off + bytes + 255) & ~(size_t)255;
        return p;
    };
    unsigned short* Xbf  = (unsigned short*)carve((size_t)MROWS * EMBED * 2);       // 16 MB
    unsigned short* WqT  = (unsigned short*)carve((size_t)3 * EMBED * EMBED * 2);   // 6 MB
    unsigned short* WoT  = (unsigned short*)carve((size_t)EMBED * EMBED * 2);       // 2 MB
    unsigned short* Qb   = (unsigned short*)carve((size_t)MROWS * EMBED * 2);       // 16 MB
    unsigned short* Kb   = (unsigned short*)carve((size_t)MROWS * EMBED * 2);       // 16 MB
    unsigned short* Vtb  = (unsigned short*)carve((size_t)MROWS * EMBED * 2);       // 16 MB
    unsigned short* Ob   = (unsigned short*)carve((size_t)MROWS * EMBED * 2);       // 16 MB

    // 1) convert inputs to bf16 (weights transposed to N-major for B-fragments)
    cvt_f32_bf16<<<(MROWS * EMBED + 255) / 256, 256, 0, stream>>>(x, Xbf, MROWS * EMBED);
    cvt_transpose_bf16<<<(EMBED * 3 * EMBED + 255) / 256, 256, 0, stream>>>(Wqkv, WqT, EMBED, 3 * EMBED);
    cvt_transpose_bf16<<<(EMBED * EMBED + 255) / 256, 256, 0, stream>>>(Wout, WoT, EMBED, EMBED);

    // 2) QKV projection: [8192x1024] @ [1024x3072], scatter to Q/K/V^T bf16
    gemm_bf16<0><<<dim3(3 * EMBED / 128, MROWS / 128), 256, 0, stream>>>(
        Xbf, WqT, bqkv, nullptr, Qb, Kb, Vtb, MROWS, 3 * EMBED, EMBED);

    // 3) flash attention: 64 (b,h) x 32 query tiles of 64 rows
    attn_kernel<<<dim3(SEQ / 64, BATCH * HEADS), 128, 0, stream>>>(Qb, Kb, Vtb, Ob);

    // 4) output projection: [8192x1024] @ [1024x1024] + bias -> fp32 out
    gemm_bf16<1><<<dim3(EMBED / 128, MROWS / 128), 256, 0, stream>>>(
        Ob, WoT, bout, out, nullptr, nullptr, nullptr, MROWS, EMBED, EMBED);
}
